// BNLSTMCell_8744553414711
// MI455X (gfx1250) — matrix operand order repaired and checked by reading, not compiled, not run on hardware
//
#include <hip/hip_runtime.h>
#include <hip/hip_bf16.h>

// BN-LSTM cell for MI455X (gfx1250).
// Dominant cost: two 4096x1024x4096 GEMMs (68.7 GFLOP) -> v_wmma_f32_16x16x32_bf16
// with f32 accumulate. bf16 operands (~34MB) are L2-resident (192MB L2).
// GEMM stages tiles into LDS (double-buffered, pad-swizzled) via the gfx1250
// async global->LDS path (ASYNCcnt) when available, giving per-block operand
// reuse so L2 traffic (~1GB) balances with WMMA compute instead of dominating.

#define BATCH 4096
#define INDIM 1024
#define HID   1024
#define FOURH 4096
#define EPSV  1e-5f

// LDS tile: 128 rows x 32 bf16, row stride padded to 40 bf16 (80B = 20 banks)
// -> conflict-free ds_load_b128 fragment reads.
#define TROWS 128
#define TK    32
#define TPAD  40

typedef __bf16  bf16_t;
typedef bf16_t  v16bf __attribute__((ext_vector_type(16)));
typedef float   v8f   __attribute__((ext_vector_type(8)));

union FragU { v16bf v; uint4 u[2]; };

#if __has_builtin(__builtin_amdgcn_global_load_async_to_lds_b128) && \
    __has_builtin(__builtin_amdgcn_s_wait_asynccnt)
#define HAVE_ASYNC_LDS 1
// The builtin's params are pointers to GNU-vector "V4i" (vector_size flavor).
typedef int v4i_gnu __attribute__((vector_size(16)));
typedef __attribute__((address_space(1))) v4i_gnu gas_v4i;   // global
typedef __attribute__((address_space(3))) v4i_gnu las_v4i;   // LDS
#else
#define HAVE_ASYNC_LDS 0
#endif

// ---------------- elementwise f32 -> bf16 convert ----------------
__global__ __launch_bounds__(256) void cvt_f32_bf16(const float* __restrict__ in,
                                                    bf16_t* __restrict__ out, int n) {
  int i = blockIdx.x * 256 + threadIdx.x;
  if (i < n) out[i] = (bf16_t)in[i];
}

// ------------- transpose + convert: W[K=1024,N=4096] -> Wt[N,K] bf16 -------------
__global__ __launch_bounds__(256) void transpose_cvt(const float* __restrict__ W,
                                                     bf16_t* __restrict__ Wt) {
  __shared__ float tile[32][33];
  const int n0 = blockIdx.x * 32;
  const int k0 = blockIdx.y * 32;
  const int tx = threadIdx.x & 31;
  const int ty = threadIdx.x >> 5;      // 32x8 threads, 4 rows each
#pragma unroll
  for (int j = 0; j < 32; j += 8)
    tile[ty + j][tx] = W[(size_t)(k0 + ty + j) * FOURH + (n0 + tx)];
  __syncthreads();
#pragma unroll
  for (int j = 0; j < 32; j += 8)
    Wt[(size_t)(n0 + ty + j) * INDIM + (k0 + tx)] = (bf16_t)tile[tx][ty + j];
}

// ---------------- bf16 WMMA GEMM with LDS-staged tiles ----------------
// C[4096,4096] = A[4096,1024] x Bt[4096,1024]^T. blockIdx.z selects GEMM 0/1.
// 8 waves/block; wave tile 32x64; block tile 128x128; K-step 32; double buffer.
__global__ __launch_bounds__(256) void gemm_bf16_wmma(
    const bf16_t* __restrict__ A0, const bf16_t* __restrict__ Bt0, float* __restrict__ C0,
    const bf16_t* __restrict__ A1, const bf16_t* __restrict__ Bt1, float* __restrict__ C1) {
  const bf16_t* __restrict__ A  = blockIdx.z ? A1  : A0;
  const bf16_t* __restrict__ Bt = blockIdx.z ? Bt1 : Bt0;
  float* __restrict__ C         = blockIdx.z ? C1  : C0;

  __shared__ __align__(16) bf16_t sA[2][TROWS * TPAD];
  __shared__ __align__(16) bf16_t sB[2][TROWS * TPAD];

  const int t    = threadIdx.x;
  const int lane = t & 31;
  const int wave = t >> 5;
  const int l15  = lane & 15;
  const int lh   = lane >> 4;                    // lane half: 0 or 1
  const int wm   = (wave & 3) * 32;              // wave row offset in block tile
  const int wn   = (wave >> 2) * 64;             // wave col offset in block tile
  const int rowBlk = blockIdx.y * TROWS;
  const int colBlk = blockIdx.x * TROWS;

  // Staging: 128 rows x 64B per tile = 512 x 16B chunks; 256 threads x 2 chunks.
  // chunk c: row = c>>2 (4 x 16B per 64B row), chunk-in-row = c&3.
  auto stage = [&](int kt, int buf) {
#pragma unroll
    for (int j = 0; j < 2; ++j) {
      const int c   = t + 256 * j;
      const int row = c >> 2;
      const int ch  = c & 3;
      const bf16_t* gA = A  + (size_t)(rowBlk + row) * INDIM + kt * TK + ch * 8;
      const bf16_t* gB = Bt + (size_t)(colBlk + row) * INDIM + kt * TK + ch * 8;
      bf16_t* lA = &sA[buf][row * TPAD + ch * 8];
      bf16_t* lB = &sB[buf][row * TPAD + ch * 8];
#if HAVE_ASYNC_LDS
      __builtin_amdgcn_global_load_async_to_lds_b128((gas_v4i*)gA, (las_v4i*)lA, 0, 0);
      __builtin_amdgcn_global_load_async_to_lds_b128((gas_v4i*)gB, (las_v4i*)lB, 0, 0);
#else
      *(uint4*)lA = *(const uint4*)gA;
      *(uint4*)lB = *(const uint4*)gB;
#endif
    }
  };

  v8f acc[2][4];
  v8f zero = {};
#pragma unroll
  for (int mt = 0; mt < 2; ++mt)
#pragma unroll
    for (int nt = 0; nt < 4; ++nt) acc[mt][nt] = zero;

  const int NT = INDIM / TK;   // 32 K-steps

  stage(0, 0);
#if HAVE_ASYNC_LDS
  __builtin_amdgcn_s_wait_asynccnt(0);
#endif
  __syncthreads();

  for (int kt = 0; kt < NT; ++kt) {
    const int buf = kt & 1;
    if (kt + 1 < NT) stage(kt + 1, buf ^ 1);   // issue next tile early

    FragU a[2], b[4];
    // A fragment (16x32 bf16): lanes 0-15 row M=l15 hold K {0..7, 16..23},
    // lanes 16-31 hold K {8..15, 24..31} -> two 16B ds_load_b128 runs.
#pragma unroll
    for (int mt = 0; mt < 2; ++mt) {
      const bf16_t* p = &sA[buf][(wm + mt * 16 + l15) * TPAD + lh * 8];
      a[mt].u[0] = *(const uint4*)(p);
      a[mt].u[1] = *(const uint4*)(p + 16);
    }
    // B fragment (32x16 bf16): lane = column N; lanes 0-15 hold K {0..7, 16..23},
    // lanes 16-31 hold K {8..15, 24..31} -> two 16B ds_load_b128 runs.
#pragma unroll
    for (int nt = 0; nt < 4; ++nt) {
      const bf16_t* p = &sB[buf][(wn + nt * 16 + l15) * TPAD + lh * 8];
      b[nt].u[0] = *(const uint4*)(p);
      b[nt].u[1] = *(const uint4*)(p + 16);
    }
#pragma unroll
    for (int mt = 0; mt < 2; ++mt)
#pragma unroll
      for (int nt = 0; nt < 4; ++nt)
        acc[mt][nt] = __builtin_amdgcn_wmma_f32_16x16x32_bf16(
            false, a[mt].v, false, b[nt].v, (short)0, acc[mt][nt], false, false);

#if HAVE_ASYNC_LDS
    __builtin_amdgcn_s_wait_asynccnt(0);       // next tile landed in LDS
#endif
    __syncthreads();
  }

  // C/D layout: VGPR v holds row (tile + lh*8 + v), col = tile + l15
#pragma unroll
  for (int mt = 0; mt < 2; ++mt) {
#pragma unroll
    for (int nt = 0; nt < 4; ++nt) {
      const int r0 = rowBlk + wm + mt * 16 + lh * 8;
      const int c  = colBlk + wn + nt * 16 + l15;
#pragma unroll
      for (int v = 0; v < 8; ++v)
        C[(size_t)(r0 + v) * FOURH + c] = acc[mt][nt][v];
    }
  }
}

// ---------------- zero the stats region ----------------
__global__ __launch_bounds__(256) void zero_f32(float* __restrict__ p, int n) {
  int i = blockIdx.x * 256 + threadIdx.x;
  if (i < n) p[i] = 0.0f;
}

// ---------------- per-column sum / sumsq over batch dim (coalesced) ----------------
__global__ __launch_bounds__(256) void col_stats(
    const float* __restrict__ wi, const float* __restrict__ wh,
    float* __restrict__ sum_i, float* __restrict__ ssq_i,
    float* __restrict__ sum_h, float* __restrict__ ssq_h) {
  const float* __restrict__ X = blockIdx.z ? wh : wi;
  float* __restrict__ S = blockIdx.z ? sum_h : sum_i;
  float* __restrict__ Q = blockIdx.z ? ssq_h : ssq_i;
  const int c  = blockIdx.x * 256 + threadIdx.x;   // column 0..4095
  const int r0 = blockIdx.y * 128;                 // 32 row-chunks
  float s = 0.f, q = 0.f;
  for (int r = r0; r < r0 + 128; ++r) {
    float v = X[(size_t)r * FOURH + c];
    s += v; q += v * v;
  }
  atomicAdd(&S[c], s);
  atomicAdd(&Q[c], q);
}

__device__ __forceinline__ float sigm(float x) { return 1.0f / (1.0f + __expf(-x)); }

// ---------------- fused BN + gates -> c1, sigma(o); accumulate c1 stats ----------------
__global__ __launch_bounds__(256) void gates_kernel(
    const float* __restrict__ wi, const float* __restrict__ wh,
    const float* __restrict__ c0, const float* __restrict__ bias,
    const float* __restrict__ g_ih, const float* __restrict__ b_ih,
    const float* __restrict__ g_hh, const float* __restrict__ b_hh,
    const float* __restrict__ sum_i, const float* __restrict__ ssq_i,
    const float* __restrict__ sum_h, const float* __restrict__ ssq_h,
    float* __restrict__ c1, float* __restrict__ sig_o,
    float* __restrict__ sum_c, float* __restrict__ ssq_c) {
  const int j  = blockIdx.x * 256 + threadIdx.x;   // hidden index 0..1023
  const int b0 = blockIdx.y * 128;                 // 32 batch-chunks
  const float invB = 1.0f / (float)BATCH;

  // Fold both batch-norms into pre = si*wi + sh*wh + off per gate column.
  float si[4], sh[4], off[4];
#pragma unroll
  for (int g = 0; g < 4; ++g) {
    const int col = j + g * HID;
    float mi = sum_i[col] * invB;
    float vi = ssq_i[col] * invB - mi * mi;
    float ai = g_ih[col] * rsqrtf(vi + EPSV);
    float mh = sum_h[col] * invB;
    float vh = ssq_h[col] * invB - mh * mh;
    float ah = g_hh[col] * rsqrtf(vh + EPSV);
    si[g]  = ai;
    sh[g]  = ah;
    off[g] = b_ih[col] - ai * mi + b_hh[col] - ah * mh + bias[col];
  }

  float s_c = 0.f, q_c = 0.f;
  for (int b = b0; b < b0 + 128; ++b) {
    const size_t base = (size_t)b * FOURH + j;
    const size_t hidx = (size_t)b * HID + j;
    float pf = si[0] * wi[base          ] + sh[0] * wh[base          ] + off[0];
    float pi = si[1] * wi[base +     HID] + sh[1] * wh[base +     HID] + off[1];
    float po = si[2] * wi[base + 2 * HID] + sh[2] * wh[base + 2 * HID] + off[2];
    float pg = si[3] * wi[base + 3 * HID] + sh[3] * wh[base + 3 * HID] + off[3];
    float cc = sigm(pf) * c0[hidx] + sigm(pi) * tanhf(pg);
    c1[hidx]    = cc;
    sig_o[hidx] = sigm(po);
    s_c += cc; q_c += cc * cc;
  }
  atomicAdd(&sum_c[j], s_c);
  atomicAdd(&ssq_c[j], q_c);
}

// ---------------- h1 = sigma(o) * tanh(bn_c(c1)) ----------------
__global__ __launch_bounds__(256) void h1_kernel(
    const float* __restrict__ c1, const float* __restrict__ sig_o,
    const float* __restrict__ g_c, const float* __restrict__ b_c,
    const float* __restrict__ sum_c, const float* __restrict__ ssq_c,
    float* __restrict__ h1) {
  const int idx = blockIdx.x * 256 + threadIdx.x;   // 0 .. BATCH*HID-1
  const int j = idx & (HID - 1);
  const float invB = 1.0f / (float)BATCH;
  float m = sum_c[j] * invB;
  float v = ssq_c[j] * invB - m * m;
  float a = g_c[j] * rsqrtf(v + EPSV);
  float bn = a * (c1[idx] - m) + b_c[j];
  h1[idx] = sig_o[idx] * tanhf(bn);
}

extern "C" void kernel_launch(void* const* d_in, const int* in_sizes, int n_in,
                              void* d_out, int out_size, void* d_ws, size_t ws_size,
                              hipStream_t stream) {
  (void)in_sizes; (void)n_in; (void)out_size; (void)ws_size;
  const float* input_   = (const float*)d_in[0];   // [4096,1024]
  const float* h_0      = (const float*)d_in[1];   // [4096,1024]
  const float* c_0      = (const float*)d_in[2];   // [4096,1024]
  const float* w_ih     = (const float*)d_in[3];   // [1024,4096]
  const float* w_hh     = (const float*)d_in[4];   // [1024,4096]
  const float* bias     = (const float*)d_in[5];   // [4096]
  const float* gamma_ih = (const float*)d_in[6];
  const float* beta_ih  = (const float*)d_in[7];
  const float* gamma_hh = (const float*)d_in[8];
  const float* beta_hh  = (const float*)d_in[9];
  const float* gamma_c  = (const float*)d_in[10];  // [1024]
  const float* beta_c   = (const float*)d_in[11];  // [1024]
  // d_in[12] = time (unused in training-mode math)

  float* h1_out = (float*)d_out;                       // [4096,1024]
  float* c1_out = (float*)d_out + (size_t)BATCH * HID; // [4096,1024]

  // ---- workspace carve-up ----
  char* w = (char*)d_ws;
  const size_t MAT = (size_t)BATCH * FOURH;            // 16,777,216
  const size_t BH  = (size_t)BATCH * HID;              //  4,194,304
  float* wi    = (float*)w;  w += MAT * sizeof(float);
  float* wh    = (float*)w;  w += MAT * sizeof(float);
  float* sig_o = (float*)w;  w += BH  * sizeof(float);
  float* stats = (float*)w;  w += (4 * FOURH + 2 * HID) * sizeof(float);
  float* sum_i = stats;
  float* ssq_i = stats + FOURH;
  float* sum_h = stats + 2 * FOURH;
  float* ssq_h = stats + 3 * FOURH;
  float* sum_c = stats + 4 * FOURH;
  float* ssq_c = stats + 4 * FOURH + HID;
  bf16_t* Abf  = (bf16_t*)w; w += BH * sizeof(bf16_t);
  bf16_t* Hbf  = (bf16_t*)w; w += BH * sizeof(bf16_t);
  bf16_t* WihT = (bf16_t*)w; w += BH * sizeof(bf16_t);
  bf16_t* WhhT = (bf16_t*)w; w += BH * sizeof(bf16_t);

  const int nStats = 4 * FOURH + 2 * HID;              // 18432
  zero_f32<<<(nStats + 255) / 256, 256, 0, stream>>>(stats, nStats);

  // f32 -> bf16 operands (L2-resident after this pass)
  cvt_f32_bf16<<<(int)(BH / 256), 256, 0, stream>>>(input_, Abf, (int)BH);
  cvt_f32_bf16<<<(int)(BH / 256), 256, 0, stream>>>(h_0,    Hbf, (int)BH);
  transpose_cvt<<<dim3(FOURH / 32, INDIM / 32), 256, 0, stream>>>(w_ih, WihT);
  transpose_cvt<<<dim3(FOURH / 32, INDIM / 32), 256, 0, stream>>>(w_hh, WhhT);

  // Both GEMMs in one launch: z=0 -> wi, z=1 -> wh
  gemm_bf16_wmma<<<dim3(FOURH / 128, BATCH / 128, 2), 256, 0, stream>>>(
      Abf, WihT, wi, Hbf, WhhT, wh);

  // per-column batch stats of wi and wh
  col_stats<<<dim3(FOURH / 256, BATCH / 128, 2), 256, 0, stream>>>(
      wi, wh, sum_i, ssq_i, sum_h, ssq_h);

  // fused BN + gates -> c1, sigma(o), and c1 stats
  gates_kernel<<<dim3(HID / 256, BATCH / 128), 256, 0, stream>>>(
      wi, wh, c_0, bias, gamma_ih, beta_ih, gamma_hh, beta_hh,
      sum_i, ssq_i, sum_h, ssq_h, c1_out, sig_o, sum_c, ssq_c);

  // h1 = sigma(o) * tanh(bn_c(c1))
  h1_kernel<<<(int)(BH / 256), 256, 0, stream>>>(
      c1_out, sig_o, gamma_c, beta_c, sum_c, ssq_c, h1_out);
}
